// PhysicsGuidedBlock_29403346108945
// MI455X (gfx1250) — matrix operand order repaired and checked by reading, not compiled, not run on hardware
//
#include <hip/hip_runtime.h>
#include <hip/hip_bf16.h>
#include <cstdint>
#include <cstddef>

// ---------------------------------------------------------------------------
// Types for gfx1250 WMMA (wave32): bf16 16x16x32, f32 accumulate
// ---------------------------------------------------------------------------
typedef __attribute__((ext_vector_type(16))) __bf16 bf16x16;
typedef __attribute__((ext_vector_type(8)))  __bf16 bf16x8;
typedef __attribute__((ext_vector_type(8)))  float  f32x8;

#define WMMA_BF16(a, b, c) \
  __builtin_amdgcn_wmma_f32_16x16x32_bf16(false, (a), false, (b), (short)0, (c), false, false)

// Problem constants
#define BB       4
#define NSEQ     2048
#define DD       512
#define HH       8
#define HD       64
#define MLP_H    2048
#define ROWS     (BB * NSEQ)                    // 8192
#define QKV_SZ   ((size_t)BB * HH * NSEQ * HD)  // 4194304 elems per q/k/v

// ---------------------------------------------------------------------------
// Async global->LDS copies (CDNA5). Builtin signature (from hipcc diagnostic):
//   void(__builtin_...b128)(v4i addrspace(1)*, v4i addrspace(3)*, imm off, imm cpol)
// Guarded so the file compiles either way; fallback is a sync 16B copy.
// ---------------------------------------------------------------------------
#if defined(__gfx1250__) && __has_builtin(__builtin_amdgcn_global_load_async_to_lds_b128)
#define HAVE_ASYNC_LDS 1
#endif

typedef int v4i_vs __attribute__((vector_size(16)));
typedef __attribute__((address_space(1))) v4i_vs* g_v4i_ptr;
typedef __attribute__((address_space(3))) v4i_vs* l_v4i_ptr;

__device__ __forceinline__ void copy16_g2l(const __bf16* __restrict__ g,
                                           __bf16* __restrict__ l) {
#ifdef HAVE_ASYNC_LDS
  __builtin_amdgcn_global_load_async_to_lds_b128((g_v4i_ptr)g, (l_v4i_ptr)l, 0, 0);
#else
  *(bf16x8*)l = *(const bf16x8*)g;
#endif
}

__device__ __forceinline__ void async_wait() {
#ifdef HAVE_ASYNC_LDS
#if __has_builtin(__builtin_amdgcn_s_wait_asynccnt)
  __builtin_amdgcn_s_wait_asynccnt(0);
#else
  asm volatile("s_wait_asynccnt 0x0" ::: "memory");
#endif
#endif
}

__device__ __forceinline__ bf16x16 combine16(bf16x8 lo, bf16x8 hi) {
  bf16x16 r;
#pragma unroll
  for (int i = 0; i < 8; ++i) { r[i] = lo[i]; r[8 + i] = hi[i]; }
  return r;
}

// ---------------------------------------------------------------------------
// Tiled transpose + fp32->bf16: src [K][N] f32  ->  dst [N][K] bf16
// ---------------------------------------------------------------------------
__global__ __launch_bounds__(256) void transpose_f2bf_kernel(
    const float* __restrict__ src, __bf16* __restrict__ dst, int K, int N) {
  __shared__ float tile[32][33];
  const int kt = blockIdx.y * 32, nt = blockIdx.x * 32;
  const int tx = threadIdx.x & 31, ty = threadIdx.x >> 5;  // 8 row-groups
#pragma unroll
  for (int r = ty; r < 32; r += 8)
    tile[r][tx] = src[(size_t)(kt + r) * N + (nt + tx)];
  __syncthreads();
#pragma unroll
  for (int r = ty; r < 32; r += 8)
    dst[(size_t)(nt + r) * K + (kt + tx)] = (__bf16)tile[tx][r];
}

// ---------------------------------------------------------------------------
// LayerNorm (D=512) -> bf16, one block per row
// ---------------------------------------------------------------------------
__global__ __launch_bounds__(256) void ln_bf16_kernel(const float* __restrict__ x,
                                                      const float* __restrict__ g,
                                                      const float* __restrict__ bta,
                                                      __bf16* __restrict__ out) {
  __shared__ float s1[256], s2[256];
  const int row = blockIdx.x, t = threadIdx.x;
  const float* xr = x + (size_t)row * DD;
  float a0 = xr[t], a1 = xr[t + 256];
  s1[t] = a0 + a1;
  s2[t] = a0 * a0 + a1 * a1;
  __syncthreads();
  for (int s = 128; s > 0; s >>= 1) {
    if (t < s) { s1[t] += s1[t + s]; s2[t] += s2[t + s]; }
    __syncthreads();
  }
  float mean = s1[0] * (1.0f / DD);
  float var  = s2[0] * (1.0f / DD) - mean * mean;
  float rstd = rsqrtf(var + 1e-5f);
  __bf16* orow = out + (size_t)row * DD;
  orow[t]       = (__bf16)((a0 - mean) * rstd * g[t]       + bta[t]);
  orow[t + 256] = (__bf16)((a1 - mean) * rstd * g[t + 256] + bta[t + 256]);
}

// ---------------------------------------------------------------------------
// bf16 WMMA GEMM: C[M,N] = A[M,K] * Bt[N,K]^T  (A row-major, B pre-transposed)
// Block tile 128x128, k-step 32, double-buffered async LDS staging.
// 8 waves (4 along M x 2 along N), wave tile 32x64 = 2x4 wmma tiles.
// MODE 0: scatter into q/k/v (q,k: [B,H,N,HD]; v: [B,H,HD,N]) bf16
// MODE 2: + bias, exact GELU, write bf16
// MODE 3: + bias + residual, write fp32
// ---------------------------------------------------------------------------
template <int MODE>
__global__ __launch_bounds__(256) void gemm_bf16_kernel(
    const __bf16* __restrict__ A, const __bf16* __restrict__ Bt,
    int M, int N, int K,
    const float* __restrict__ bias, const float* __restrict__ resid,
    __bf16* __restrict__ out_bf, float* __restrict__ out_f) {
  __shared__ __attribute__((aligned(16))) __bf16 As [2][128][40];  // [m][k]
  __shared__ __attribute__((aligned(16))) __bf16 Bts[2][128][40];  // [n][k]

  const int t = threadIdx.x;
  const int lane = t & 31, wid = t >> 5;
  const int wm = (wid >> 1) * 32;
  const int wn = (wid & 1) * 64;
  const int lr = lane & 15, lh = lane >> 4;
  const int n0 = blockIdx.x * 128;
  const int m0 = blockIdx.y * 128;

  f32x8 acc[2][4];
#pragma unroll
  for (int i = 0; i < 2; ++i)
#pragma unroll
    for (int j = 0; j < 4; ++j)
#pragma unroll
      for (int e = 0; e < 8; ++e) acc[i][j][e] = 0.0f;

  auto stage = [&](int k0, int buf) {
    // A tile 128x32 and Bt tile 128x32: 512 chunks of 8 bf16 each
#pragma unroll
    for (int i = 0; i < 2; ++i) {
      int c = t + 256 * i;
      int r = c >> 2, kc = (c & 3) * 8;
      copy16_g2l(A + (size_t)(m0 + r) * K + (k0 + kc), &As[buf][r][kc]);
    }
#pragma unroll
    for (int i = 0; i < 2; ++i) {
      int c = t + 256 * i;
      int r = c >> 2, kc = (c & 3) * 8;
      copy16_g2l(Bt + (size_t)(n0 + r) * K + (k0 + kc), &Bts[buf][r][kc]);
    }
  };

  const int KT = K >> 5;
  stage(0, 0);
  async_wait();
  __syncthreads();

  int buf = 0;
  for (int kt = 0; kt < KT; ++kt) {
    if (kt + 1 < KT) stage((kt + 1) << 5, buf ^ 1);

    bf16x16 af[2];
#pragma unroll
    for (int mi = 0; mi < 2; ++mi) {
      const __bf16* p = &As[buf][wm + mi * 16 + lr][lh * 8];
      af[mi] = combine16(*(const bf16x8*)p, *(const bf16x8*)(p + 16));
    }
#pragma unroll
    for (int ni = 0; ni < 4; ++ni) {
      const __bf16* p = &Bts[buf][wn + ni * 16 + lr][lh * 8];
      bf16x16 bfr = combine16(*(const bf16x8*)p, *(const bf16x8*)(p + 16));
#pragma unroll
      for (int mi = 0; mi < 2; ++mi)
        acc[mi][ni] = WMMA_BF16(af[mi], bfr, acc[mi][ni]);
    }
    async_wait();
    __syncthreads();
    buf ^= 1;
  }

  // Epilogue. C layout: vgpr r -> row = 8*lh + r (+tile), col = lr (+tile)
#pragma unroll
  for (int mi = 0; mi < 2; ++mi)
#pragma unroll
    for (int ni = 0; ni < 4; ++ni)
#pragma unroll
      for (int r = 0; r < 8; ++r) {
        int row = m0 + wm + mi * 16 + lh * 8 + r;
        int col = n0 + wn + ni * 16 + lr;
        float val = acc[mi][ni][r];
        if (MODE == 0) {
          int which = col >> 9, cc = col & 511;
          int hh = cc >> 6, dd2 = cc & 63;
          int bbx = row >> 11, nn = row & 2047;
          size_t idx;
          if (which == 2)  // V stored transposed [B,H,HD,N]
            idx = 2 * QKV_SZ + ((((size_t)bbx * HH + hh) * HD + dd2) * NSEQ + nn);
          else
            idx = (size_t)which * QKV_SZ + ((((size_t)bbx * HH + hh) * NSEQ + nn) * HD + dd2);
          out_bf[idx] = (__bf16)val;
        } else if (MODE == 2) {
          float xg = val + bias[col];
          float ge = 0.5f * xg * (1.0f + erff(xg * 0.70710678118654752f));
          out_bf[(size_t)row * N + col] = (__bf16)ge;
        } else {
          out_f[(size_t)row * N + col] =
              val + bias[col] + resid[(size_t)row * N + col];
        }
      }
}

// ---------------------------------------------------------------------------
// Flash attention, one block per (b, h, 64-row tile). 8 waves.
// q/k: bf16 [B,H,N,HD]; v: bf16 [B,H,HD,N] (pre-transposed by QKV epilogue)
// out: bf16 [B,N,D] (heads re-gathered). Double-buffered async K/V staging.
// ---------------------------------------------------------------------------
__global__ __launch_bounds__(256) void flash_attn_kernel(
    const __bf16* __restrict__ q, const __bf16* __restrict__ k,
    const __bf16* __restrict__ vt, const float* __restrict__ elev,
    const float* __restrict__ alpha, __bf16* __restrict__ out) {
  const int mt = blockIdx.x, h = blockIdx.y, b = blockIdx.z;
  const int m0 = mt * 64;
  const size_t bh = ((size_t)b * HH + h) * NSEQ * HD;
  const __bf16* qbh  = q  + bh;
  const __bf16* kbh  = k  + bh;
  const __bf16* vtbh = vt + bh;  // [HD][NSEQ]
  const float* ev = elev + (size_t)b * NSEQ;

  __shared__ __attribute__((aligned(16))) __bf16 Qs[64][72];
  __shared__ __attribute__((aligned(16))) __bf16 Ks[2][64][72];   // [key][d]
  __shared__ __attribute__((aligned(16))) __bf16 Vts[2][64][72];  // [d][key]
  __shared__ __attribute__((aligned(16))) __bf16 Ps[64][72];
  __shared__ __attribute__((aligned(16))) float  Ss[64][68];
  __shared__ float mrow[64], lrow[64], srow[64];
  __shared__ float er_s[64], ec_s[2][64];

  const int t = threadIdx.x;
  const int lane = t & 31, wid = t >> 5;
  const int wm = (wid >> 1) * 16;   // 4 waves along rows
  const int wn = (wid & 1) * 32;    // 2 waves along cols
  const int lr = lane & 15, lh = lane >> 4;
  const float al = alpha[0];

  auto stage_kv = [&](int j0, int buf) {
#pragma unroll
    for (int i = 0; i < 2; ++i) {
      int c = t + 256 * i;
      int r = c >> 3, dc = (c & 7) * 8;
      copy16_g2l(kbh  + (size_t)(j0 + r) * HD + dc,   &Ks [buf][r][dc]);
      copy16_g2l(vtbh + (size_t)r * NSEQ + j0 + dc,   &Vts[buf][r][dc]);
    }
    if (t < 64) ec_s[buf][t] = ev[j0 + t];
  };

  // Stage Q (once) + first K/V tile
#pragma unroll
  for (int i = 0; i < 2; ++i) {
    int c = t + 256 * i;
    int r = c >> 3, dc = (c & 7) * 8;
    copy16_g2l(qbh + (size_t)(m0 + r) * HD + dc, &Qs[r][dc]);
  }
  stage_kv(0, 0);
  if (t < 64) { er_s[t] = ev[m0 + t]; mrow[t] = -1e30f; lrow[t] = 0.0f; }

  f32x8 oacc[2];
#pragma unroll
  for (int i = 0; i < 2; ++i)
#pragma unroll
    for (int e = 0; e < 8; ++e) oacc[i][e] = 0.0f;

  async_wait();
  __syncthreads();

  int buf = 0;
  for (int jt = 0; jt < NSEQ / 64; ++jt) {
    if (jt + 1 < NSEQ / 64) stage_kv((jt + 1) * 64, buf ^ 1);

    // S tile: rows [wm,wm+16) x cols [wn,wn+32), K-dim = 64
    f32x8 sacc[2];
#pragma unroll
    for (int i = 0; i < 2; ++i)
#pragma unroll
      for (int e = 0; e < 8; ++e) sacc[i][e] = 0.0f;
#pragma unroll
    for (int ks = 0; ks < 2; ++ks) {
      const int kb = ks * 32;
      const __bf16* ap = &Qs[wm + lr][kb + lh * 8];
      bf16x16 af = combine16(*(const bf16x8*)ap, *(const bf16x8*)(ap + 16));
#pragma unroll
      for (int ni = 0; ni < 2; ++ni) {
        const __bf16* bp = &Ks[buf][wn + ni * 16 + lr][kb + lh * 8];
        bf16x16 bfr = combine16(*(const bf16x8*)bp, *(const bf16x8*)(bp + 16));
        sacc[ni] = WMMA_BF16(af, bfr, sacc[ni]);
      }
    }
#pragma unroll
    for (int ni = 0; ni < 2; ++ni)
#pragma unroll
      for (int r = 0; r < 8; ++r)
        Ss[wm + lh * 8 + r][wn + ni * 16 + lr] = sacc[ni][r];
    __syncthreads();

    // Online softmax: 256 threads, 4 lanes per row, shfl_xor reductions
    {
      const int r = t >> 2, qq = t & 3;
      const int c0 = qq * 16;
      const float er = er_s[r];
      float sv[16];
      float mx = -3.0e30f;
#pragma unroll
      for (int j = 0; j < 16; ++j) {
        float ed = (ec_s[buf][c0 + j] - er) * 1e-3f;
        float bb2 = fmaxf(-al * fmaxf(ed, 0.0f), -10.0f);
        sv[j] = Ss[r][c0 + j] * 0.125f + bb2;  // SCALE = 64^-0.5
        mx = fmaxf(mx, sv[j]);
      }
      mx = fmaxf(mx, __shfl_xor(mx, 1, 32));
      mx = fmaxf(mx, __shfl_xor(mx, 2, 32));
      const float mold = mrow[r];
      const float mnew = fmaxf(mold, mx);
      float ls = 0.0f;
#pragma unroll
      for (int j = 0; j < 16; ++j) {
        float p = __expf(sv[j] - mnew);
        ls += p;
        Ps[r][c0 + j] = (__bf16)p;
      }
      ls += __shfl_xor(ls, 1, 32);
      ls += __shfl_xor(ls, 2, 32);
      if (qq == 0) {
        float esc = __expf(mold - mnew);
        lrow[r] = lrow[r] * esc + ls;
        mrow[r] = mnew;
        srow[r] = esc;
      }
    }
    __syncthreads();

    // Rescale running O, then O += P (64x64) @ V (64x64)
#pragma unroll
    for (int ni = 0; ni < 2; ++ni)
#pragma unroll
      for (int r = 0; r < 8; ++r) oacc[ni][r] *= srow[wm + lh * 8 + r];
#pragma unroll
    for (int ks = 0; ks < 2; ++ks) {
      const int kb = ks * 32;
      const __bf16* ap = &Ps[wm + lr][kb + lh * 8];
      bf16x16 af = combine16(*(const bf16x8*)ap, *(const bf16x8*)(ap + 16));
#pragma unroll
      for (int ni = 0; ni < 2; ++ni) {
        const __bf16* bp = &Vts[buf][wn + ni * 16 + lr][kb + lh * 8];
        bf16x16 bfr = combine16(*(const bf16x8*)bp, *(const bf16x8*)(bp + 16));
        oacc[ni] = WMMA_BF16(af, bfr, oacc[ni]);
      }
    }
    async_wait();
    __syncthreads();   // protect Ss/Ps + buffer swap
    buf ^= 1;
  }

  // Normalize rows by l, gather heads into [B,N,D]
#pragma unroll
  for (int ni = 0; ni < 2; ++ni)
#pragma unroll
    for (int r = 0; r < 8; ++r) {
      int lrw = wm + lh * 8 + r;
      int dcol = wn + ni * 16 + lr;
      float val = oacc[ni][r] / lrow[lrw];
      out[(((size_t)b * NSEQ + (m0 + lrw)) * DD) + h * HD + dcol] = (__bf16)val;
    }
}

// ---------------------------------------------------------------------------
// Launcher
// ---------------------------------------------------------------------------
extern "C" void kernel_launch(void* const* d_in, const int* in_sizes, int n_in,
                              void* d_out, int out_size, void* d_ws, size_t ws_size,
                              hipStream_t stream) {
  const float* x      = (const float*)d_in[0];
  const float* elev   = (const float*)d_in[1];
  const float* ln1_g  = (const float*)d_in[2];
  const float* ln1_b  = (const float*)d_in[3];
  const float* qkv_w  = (const float*)d_in[4];
  const float* alpha  = (const float*)d_in[5];
  const float* proj_w = (const float*)d_in[6];
  const float* proj_b = (const float*)d_in[7];
  const float* ln2_g  = (const float*)d_in[8];
  const float* ln2_b  = (const float*)d_in[9];
  const float* fc1_w  = (const float*)d_in[10];
  const float* fc1_b  = (const float*)d_in[11];
  const float* fc2_w  = (const float*)d_in[12];
  const float* fc2_b  = (const float*)d_in[13];

  char* w = (char*)d_ws;
  __bf16* wqkv  = (__bf16*)w; w += (size_t)DD * 3 * DD * 2;   // [3D][D] transposed
  __bf16* wproj = (__bf16*)w; w += (size_t)DD * DD * 2;       // [D][D] transposed
  __bf16* wfc1  = (__bf16*)w; w += (size_t)DD * MLP_H * 2;    // [MLP_H][D] transposed
  __bf16* wfc2  = (__bf16*)w; w += (size_t)MLP_H * DD * 2;    // [D][MLP_H] transposed
  __bf16* h1    = (__bf16*)w; w += (size_t)ROWS * DD * 2;
  __bf16* qkv   = (__bf16*)w; w += 3 * QKV_SZ * 2;
  __bf16* attn  = (__bf16*)w; w += (size_t)ROWS * DD * 2;
  float*  x1    = (float*) w; w += (size_t)ROWS * DD * 4;
  __bf16* h2    = (__bf16*)w; w += (size_t)ROWS * DD * 2;
  __bf16* a1    = (__bf16*)w; w += (size_t)ROWS * MLP_H * 2;

  // Weight conversion + transpose: [K][N] f32 -> [N][K] bf16
  transpose_f2bf_kernel<<<dim3((3 * DD) / 32, DD / 32), 256, 0, stream>>>(qkv_w, wqkv, DD, 3 * DD);
  transpose_f2bf_kernel<<<dim3(DD / 32, DD / 32), 256, 0, stream>>>(proj_w, wproj, DD, DD);
  transpose_f2bf_kernel<<<dim3(MLP_H / 32, DD / 32), 256, 0, stream>>>(fc1_w, wfc1, DD, MLP_H);
  transpose_f2bf_kernel<<<dim3(DD / 32, MLP_H / 32), 256, 0, stream>>>(fc2_w, wfc2, MLP_H, DD);

  // LN1
  ln_bf16_kernel<<<ROWS, 256, 0, stream>>>(x, ln1_g, ln1_b, h1);

  // QKV GEMM (scatter: q,k -> [B,H,N,HD]; v -> [B,H,HD,N])
  gemm_bf16_kernel<0><<<dim3((3 * DD) / 128, ROWS / 128), 256, 0, stream>>>(
      h1, wqkv, ROWS, 3 * DD, DD, nullptr, nullptr, qkv, nullptr);

  // Flash attention
  flash_attn_kernel<<<dim3(NSEQ / 64, HH, BB), 256, 0, stream>>>(
      qkv, qkv + QKV_SZ, qkv + 2 * QKV_SZ, elev, alpha, attn);

  // proj + bias + residual -> x1 (fp32)
  gemm_bf16_kernel<3><<<dim3(DD / 128, ROWS / 128), 256, 0, stream>>>(
      attn, wproj, ROWS, DD, DD, proj_b, x, nullptr, x1);

  // LN2
  ln_bf16_kernel<<<ROWS, 256, 0, stream>>>(x1, ln2_g, ln2_b, h2);

  // fc1 + bias + GELU -> a1 (bf16)
  gemm_bf16_kernel<2><<<dim3(MLP_H / 128, ROWS / 128), 256, 0, stream>>>(
      h2, wfc1, ROWS, MLP_H, DD, fc1_b, nullptr, a1, nullptr);

  // fc2 + bias + residual(x1) -> d_out (fp32)
  gemm_bf16_kernel<3><<<dim3(DD / 128, ROWS / 128), 256, 0, stream>>>(
      a1, wfc2, ROWS, DD, MLP_H, fc2_b, x1, nullptr, (float*)d_out);
}
